// QMLPEncoder_84172769067210
// MI455X (gfx1250) — compile-verified, matrix operand order checked by reading, not run on hardware
//
#include <hip/hip_runtime.h>
#include <hip/hip_bf16.h>

typedef __attribute__((ext_vector_type(16))) _Float16 v16h;
typedef __attribute__((ext_vector_type(8)))  _Float16 v8h;
typedef __attribute__((ext_vector_type(8)))  float    v8f;

#define NQ 4
#define HID 256
#define IN_DIM 512
#define LDA 264   // padded f16 row stride for LDS tile (row = 528B, 16B aligned)

// ---------------------------------------------------------------------------
// Kernel 1: transpose + convert W2/W3 (f32 [K][N]) -> f16 [N][K] in workspace
// ---------------------------------------------------------------------------
__global__ void k_convert(const float* __restrict__ W2, const float* __restrict__ W3,
                          _Float16* __restrict__ W2T, _Float16* __restrict__ W3T) {
    int idx = blockIdx.x * 256 + threadIdx.x;       // 0 .. 131071
    int sel = idx >> 16;
    int e   = idx & 65535;
    int n   = e >> 8;
    int k   = e & 255;
    const float* W = sel ? W3 : W2;
    _Float16*    T = sel ? W3T : W2T;
    T[n * 256 + k] = (_Float16)W[k * 256 + n];
}

// ---------------------------------------------------------------------------
// Kernel 2: h4 = relu(x @ W1[:, 0:4] + b1[0:4])   (only 4 cols of h are live)
// ---------------------------------------------------------------------------
__global__ void k_linear4(const float* __restrict__ x, const float* __restrict__ W1,
                          const float* __restrict__ b1, float* __restrict__ h4) {
    __shared__ float4 wcol[IN_DIM];                 // 8KB: cols 0..3 of each W1 row
    int tid = threadIdx.x;
    for (int i = tid; i < IN_DIM; i += 256)
        wcol[i] = *(const float4*)&W1[(long)i * HID];   // contiguous 4 floats
    __syncthreads();

    long row = (long)blockIdx.x * 256 + tid;
    const float4* xr = (const float4*)(x + row * IN_DIM);
    float a0 = b1[0], a1 = b1[1], a2 = b1[2], a3 = b1[3];
    #pragma unroll 4
    for (int k4 = 0; k4 < IN_DIM / 4; ++k4) {
        float4 xv = xr[k4];
        float4 w0 = wcol[k4 * 4 + 0];
        float4 w1 = wcol[k4 * 4 + 1];
        float4 w2 = wcol[k4 * 4 + 2];
        float4 w3 = wcol[k4 * 4 + 3];
        a0 = fmaf(xv.x, w0.x, fmaf(xv.y, w1.x, fmaf(xv.z, w2.x, fmaf(xv.w, w3.x, a0))));
        a1 = fmaf(xv.x, w0.y, fmaf(xv.y, w1.y, fmaf(xv.z, w2.y, fmaf(xv.w, w3.y, a1))));
        a2 = fmaf(xv.x, w0.z, fmaf(xv.y, w1.z, fmaf(xv.z, w2.z, fmaf(xv.w, w3.z, a2))));
        a3 = fmaf(xv.x, w0.w, fmaf(xv.y, w1.w, fmaf(xv.z, w2.w, fmaf(xv.w, w3.w, a3))));
    }
    float4 r;
    r.x = fmaxf(a0, 0.f); r.y = fmaxf(a1, 0.f);
    r.z = fmaxf(a2, 0.f); r.w = fmaxf(a3, 0.f);
    *(float4*)&h4[row * 4] = r;
}

// ---------------------------------------------------------------------------
// Kernel 3: 4-qubit statevector sim per sample + readout relu(z@Wr+br) -> h1
// Qubit q lives on bit (3-q) of the state index.
// ---------------------------------------------------------------------------
__global__ void k_quantum(const float* __restrict__ h4, const float* __restrict__ qp,
                          const float* __restrict__ Wr, const float* __restrict__ br,
                          float* __restrict__ h1) {
    __shared__ float zbuf[256][4];                  // 4KB
    int tid = threadIdx.x;
    long s0 = (long)blockIdx.x * 256;

    float4 ang = *(const float4*)&h4[(s0 + tid) * 4];
    float angs[4] = {ang.x, ang.y, ang.z, ang.w};

    float sr[16], si[16];
    #pragma unroll
    for (int i = 0; i < 16; ++i) { sr[i] = 0.f; si[i] = 0.f; }
    sr[0] = 1.f;

    // Angle embedding: RY(angle_q) on qubit q (real gate)
    #pragma unroll
    for (int q = 0; q < 4; ++q) {
        float th = angs[q] * 0.5f;
        float c = cosf(th), s = sinf(th);
        int m = 1 << (3 - q);
        #pragma unroll
        for (int i = 0; i < 16; ++i) {
            if (!(i & m)) {
                int j = i | m;
                float r0 = sr[i], i0 = si[i], r1 = sr[j], i1 = si[j];
                sr[i] = c * r0 - s * r1;  si[i] = c * i0 - s * i1;
                sr[j] = s * r0 + c * r1;  si[j] = s * i0 + c * i1;
            }
        }
    }
    // Rot(phi,theta,omega) = RZ(omega) RY(theta) RZ(phi) per qubit (shared weights)
    #pragma unroll
    for (int q = 0; q < 4; ++q) {
        float phi = qp[q * 3 + 0], th = qp[q * 3 + 1], om = qp[q * 3 + 2];
        float c = cosf(th * 0.5f), s = sinf(th * 0.5f);
        float ap = (phi + om) * 0.5f, am = (phi - om) * 0.5f;
        float epr = cosf(ap), epi = -sinf(ap);
        float emr = cosf(am), emi = -sinf(am);
        float g00r =  epr * c, g00i =  epi * c;
        float g01r = -emr * s, g01i =  emi * s;
        float g10r =  emr * s, g10i =  emi * s;
        float g11r =  epr * c, g11i = -epi * c;
        int m = 1 << (3 - q);
        #pragma unroll
        for (int i = 0; i < 16; ++i) {
            if (!(i & m)) {
                int j = i | m;
                float r0 = sr[i], i0 = si[i], r1 = sr[j], i1 = si[j];
                sr[i] = g00r * r0 - g00i * i0 + g01r * r1 - g01i * i1;
                si[i] = g00r * i0 + g00i * r0 + g01r * i1 + g01i * r1;
                sr[j] = g10r * r0 - g10i * i0 + g11r * r1 - g11i * i1;
                si[j] = g10r * i0 + g10i * r0 + g11r * i1 + g11i * r1;
            }
        }
    }
    // CNOT chain: control q (bit 3-q), target q+1 (bit 2-q)
    #pragma unroll
    for (int q = 0; q < 3; ++q) {
        int cm = 1 << (3 - q), tm = 1 << (2 - q);
        #pragma unroll
        for (int i = 0; i < 16; ++i) {
            if ((i & cm) && !(i & tm)) {
                int j = i | tm;
                float tr = sr[i]; sr[i] = sr[j]; sr[j] = tr;
                float ti = si[i]; si[i] = si[j]; si[j] = ti;
            }
        }
    }
    // <Z_q>
    #pragma unroll
    for (int q = 0; q < 4; ++q) {
        int m = 1 << (3 - q);
        float acc = 0.f;
        #pragma unroll
        for (int i = 0; i < 16; ++i) {
            float p = sr[i] * sr[i] + si[i] * si[i];
            acc += (i & m) ? -p : p;
        }
        zbuf[tid][q] = acc;
    }
    __syncthreads();

    // Readout: thread owns column n = tid; coalesced stores across samples.
    float w0 = Wr[0 * HID + tid], w1 = Wr[1 * HID + tid];
    float w2 = Wr[2 * HID + tid], w3 = Wr[3 * HID + tid];
    float bb = br[tid];
    for (int s = 0; s < 256; ++s) {
        float4 zv = *(const float4*)&zbuf[s][0];
        float v = fmaf(zv.x, w0, fmaf(zv.y, w1, fmaf(zv.z, w2, fmaf(zv.w, w3, bb))));
        h1[(s0 + s) * HID + tid] = fmaxf(v, 0.f);
    }
}

// ---------------------------------------------------------------------------
// Kernel 4: fused tail: out = relu(h1@W2+b2) @ W3 + b3  (WMMA f16, f32 accum)
// Block = 256 threads = 8 waves; 64 rows x 256 cols per block.
// Wave (w&3) -> M-tile, (w>>2) -> N-half of 128. 8 C-tiles per wave.
// ---------------------------------------------------------------------------
__device__ inline v16h frag_lds(const _Float16* base, int row16, int k0, int lane) {
    // 16-bit A-matrix layout: lanes 0-15 hold K k0..k0+7 / k0+16..k0+23;
    // lanes 16-31 hold K k0+8..k0+15 / k0+24..k0+31 (row = lane & 15).
    int r  = row16 + (lane & 15);
    int ko = k0 + ((lane >> 4) << 3);
    const _Float16* p = base + r * LDA + ko;
    union { v16h v; v8h h[2]; } u;
    u.h[0] = *(const v8h*)p;
    u.h[1] = *(const v8h*)(p + 16);
    return u.v;
}

__device__ inline v16h frag_gbl(const _Float16* WT, int n16, int k0, int lane) {
    // WT is [N][K] row-major f16; same per-lane K pattern, N takes row role.
    int r  = n16 + (lane & 15);
    int ko = k0 + ((lane >> 4) << 3);
    const _Float16* p = WT + r * 256 + ko;
    union { v16h v; v8h h[2]; } u;
    u.h[0] = *(const v8h*)p;
    u.h[1] = *(const v8h*)(p + 16);
    return u.v;
}

__global__ void k_mlp_tail(const _Float16* __restrict__ W2T, const _Float16* __restrict__ W3T,
                           const float* __restrict__ b2, const float* __restrict__ b3,
                           float* __restrict__ io /* h1 in, final out (in-place per block) */) {
    __shared__ __align__(16) _Float16 A[64 * LDA];  // ~33.8 KB
    int tid  = threadIdx.x;
    int lane = tid & 31;
    int wave = tid >> 5;
    long rb  = (long)blockIdx.x * 64;

    // Stage h1 tile as f16
    for (int i = tid; i < 64 * 256; i += 256) {
        int r = i >> 8, c = i & 255;
        A[r * LDA + c] = (_Float16)io[(rb + r) * HID + c];
    }
    __syncthreads();

    int m0 = (wave & 3) << 4;    // 0,16,32,48
    int n0 = (wave >> 2) << 7;   // 0,128

    v8f acc[8];
    #pragma unroll
    for (int t = 0; t < 8; ++t) acc[t] = v8f{};

    // GEMM 1: h2 = h1 @ W2
    for (int k0 = 0; k0 < 256; k0 += 32) {
        v16h a = frag_lds(A, m0, k0, lane);
        #pragma unroll
        for (int t = 0; t < 8; ++t) {
            v16h b = frag_gbl(W2T, n0 + (t << 4), k0, lane);
            acc[t] = __builtin_amdgcn_wmma_f32_16x16x32_f16(
                false, a, false, b, (short)0, acc[t], false, false);
        }
    }
    __syncthreads();   // all waves done reading h1 tile

    // Epilogue 1: +b2, ReLU, write h2 back into LDS as f16
    int lm = (lane >> 4) << 3;   // 0 or 8
    int ln = lane & 15;
    #pragma unroll
    for (int t = 0; t < 8; ++t) {
        int N = n0 + (t << 4) + ln;
        float bv = b2[N];
        #pragma unroll
        for (int r = 0; r < 8; ++r) {
            // C layout: VGPR r, lanes 0-15 -> M=r; lanes 16-31 -> M=8+r; N=lane&15
            float v = acc[t][r] + bv;
            A[(m0 + lm + r) * LDA + N] = (_Float16)fmaxf(v, 0.f);
        }
        acc[t] = v8f{};
    }
    __syncthreads();   // h2 tile complete

    // GEMM 2: out = h2 @ W3
    for (int k0 = 0; k0 < 256; k0 += 32) {
        v16h a = frag_lds(A, m0, k0, lane);
        #pragma unroll
        for (int t = 0; t < 8; ++t) {
            v16h b = frag_gbl(W3T, n0 + (t << 4), k0, lane);
            acc[t] = __builtin_amdgcn_wmma_f32_16x16x32_f16(
                false, a, false, b, (short)0, acc[t], false, false);
        }
    }

    // Epilogue 2: +b3, f32 store (block writes only its own rows -> in-place ok)
    #pragma unroll
    for (int t = 0; t < 8; ++t) {
        int N = n0 + (t << 4) + ln;
        float bv = b3[N];
        #pragma unroll
        for (int r = 0; r < 8; ++r) {
            io[(rb + m0 + lm + r) * HID + N] = acc[t][r] + bv;
        }
    }
}

// ---------------------------------------------------------------------------
extern "C" void kernel_launch(void* const* d_in, const int* in_sizes, int n_in,
                              void* d_out, int out_size, void* d_ws, size_t ws_size,
                              hipStream_t stream) {
    const float* x   = (const float*)d_in[0];   // [B, 512]
    const float* W1  = (const float*)d_in[1];   // [512, 256]
    const float* b1  = (const float*)d_in[2];   // [256]
    const float* qpp = (const float*)d_in[3];   // [1, 4, 3]
    const float* Wr  = (const float*)d_in[4];   // [4, 256]
    const float* br  = (const float*)d_in[5];   // [256]
    const float* W2  = (const float*)d_in[6];   // [256, 256]
    const float* b2  = (const float*)d_in[7];   // [256]
    const float* W3  = (const float*)d_in[8];   // [256, 256]
    const float* b3  = (const float*)d_in[9];   // [256]
    float* out = (float*)d_out;                 // [B, 256]

    const int B = in_sizes[0] / IN_DIM;         // 131072

    // Workspace layout: W2T f16 (128KB) | W3T f16 (128KB) | h4 f32 (B*4*4)
    _Float16* W2T = (_Float16*)d_ws;
    _Float16* W3T = W2T + 256 * 256;
    float*    h4  = (float*)(W3T + 256 * 256);

    k_convert<<<512, 256, 0, stream>>>(W2, W3, W2T, W3T);
    k_linear4<<<B / 256, 256, 0, stream>>>(x, W1, b1, h4);
    k_quantum<<<B / 256, 256, 0, stream>>>(h4, qpp, Wr, br, out);
    k_mlp_tail<<<B / 64, 256, 0, stream>>>(W2T, W3T, b2, b3, out);
}
